// dclstm_16922171146514
// MI455X (gfx1250) — compile-verified, matrix operand order checked by reading, not run on hardware
//
#include <hip/hip_runtime.h>
#include <hip/hip_bf16.h>
#include <math.h>

typedef __attribute__((ext_vector_type(2)))  float    v2f;
typedef __attribute__((ext_vector_type(8)))  float    v8f;
typedef __attribute__((ext_vector_type(16))) _Float16 v16h;

#define TLEN 256
// K dimension of the final projection padded to 64 (covers both 13x4 and 2x32 tiling)
#define KPAD 64

// ---------------------------------------------------------------------------
// Kernel 1: one wave (32 lanes) builds the 16x16 output lookup table.
//   pre[v][f] = emb[v] . w1[f] + b1[f] + b2[f]        (16 x 200)
//   h[v][j]   = sigmoid(pre[..,100+j]) * tanh( sigmoid(pre[..,j]) * tanh(pre[..,150+j]) )
//   table[v][n] = sum_k h[v][k] * w_out[n][k] + b_out[n]   <-- WMMA f32 16x16xK
// ---------------------------------------------------------------------------
__global__ __launch_bounds__(32) void dclstm_build_table(
    const float* __restrict__ emb, const float* __restrict__ w1,
    const float* __restrict__ b1,  const float* __restrict__ b2,
    const float* __restrict__ w_out, const float* __restrict__ b_out,
    float* __restrict__ table)
{
    __shared__ float pre[16][200];
    __shared__ float hA[16][KPAD];   // A-matrix (16 x K), zero-padded
    __shared__ float wB[KPAD][16];   // B-matrix (K x 16) = w_out^T, zero-padded

    const int lane = threadIdx.x;

    // preact: 3200 dot products of length 10 (100 per lane, no divergence)
    for (int t = lane; t < 16 * 200; t += 32) {
        int v = t / 200, f = t % 200;
        float acc = b1[f] + b2[f];
#pragma unroll
        for (int e = 0; e < 10; ++e) acc += emb[v * 10 + e] * w1[f * 10 + e];
        pre[v][f] = acc;
    }
    // zero the padded fragments
    for (int t = lane; t < 16 * KPAD; t += 32) hA[t / KPAD][t % KPAD] = 0.0f;
    for (int t = lane; t < KPAD * 16; t += 32) wB[t / 16][t % 16] = 0.0f;
    __syncthreads();

    // gates -> h  (800 elements, 25 per lane, uniform trip count)
    for (int t = lane; t < 16 * 50; t += 32) {
        int v = t / 50, j = t % 50;
        float i_g = 1.0f / (1.0f + expf(-pre[v][j]));
        float o_g = 1.0f / (1.0f + expf(-pre[v][100 + j]));
        float g   = tanhf(pre[v][150 + j]);
        float c   = i_g * g;
        hA[v][j]  = o_g * tanhf(c);
    }
    // B = w_out^T  (wB[k][n] = w_out[n][k])
    for (int t = lane; t < 16 * 50; t += 32) {
        int n = t / 50, k = t % 50;
        wB[k][n] = w_out[n * 50 + k];
    }
    __syncthreads();

    // ---- WMMA: D(16x16) = A(16xK) * B(Kx16), accumulate in f32 -------------
    const int m  = lane & 15;   // A row / B col / D col (per ISA lane layout)
    const int kh = lane >> 4;   // which K-half this lane holds
    v8f c = {};

#if __has_builtin(__builtin_amdgcn_wmma_f32_16x16x4_f32)
    // f32 path: 13 k-steps of 4 (K padded 50->52), exact fp32 accumulation.
    // A 16x4 layout: lanes 0-15 hold K={0,1}, lanes 16-31 hold K={2,3}.
#pragma unroll
    for (int kc = 0; kc < 13; ++kc) {
        int k0 = kc * 4 + kh * 2;
        v2f a, b;
        a.x = hA[m][k0];      a.y = hA[m][k0 + 1];
        b.x = wB[k0][m];      b.y = wB[k0 + 1][m];
        c = __builtin_amdgcn_wmma_f32_16x16x4_f32(
                /*neg_a=*/false, a, /*neg_b=*/false, b,
                /*c_mod=*/(short)0, c, /*reuse_a=*/false, /*reuse_b=*/false);
    }
#else
    // Fallback: codegen-confirmed f16 WMMA, K padded 50->64, 2 k-chunks of 32.
    // 16-bit A 16x32 layout: element i of v16h maps to K = (i&8)*2 + kh*8 + (i&7).
#pragma unroll
    for (int kc = 0; kc < 2; ++kc) {
        v16h a, b;
#pragma unroll
        for (int i = 0; i < 16; ++i) {
            int k = kc * 32 + ((i & 8) * 2) + kh * 8 + (i & 7);
            a[i] = (_Float16)hA[m][k];
            b[i] = (_Float16)wB[k][m];
        }
        c = __builtin_amdgcn_wmma_f32_16x16x32_f16(
                false, a, false, b, (short)0, c, false, false);
    }
#endif

    // D layout: VGPR r, lanes 0-15 -> M=r, lanes 16-31 -> M=r+8; N = lane&15
    const float bo = b_out[m];
#pragma unroll
    for (int r = 0; r < 8; ++r)
        table[(r + kh * 8) * 16 + m] = c[r] + bo;
}

// ---------------------------------------------------------------------------
// Kernel 2: out[b, :] = table[x[b, T-1], :]   (one thread per batch row)
// ---------------------------------------------------------------------------
__global__ __launch_bounds__(256) void dclstm_gather(
    const int* __restrict__ x, const float* __restrict__ table,
    float* __restrict__ out)
{
    __shared__ float4 t4[64];                 // 16 rows x 16 floats
    const int lt = threadIdx.x;
    if (lt < 64) t4[lt] = reinterpret_cast<const float4*>(table)[lt];
    __syncthreads();

    const int b   = blockIdx.x * 256 + lt;    // 16 blocks x 256 = 4096 rows
    const int idx = x[b * TLEN + (TLEN - 1)] & 15;

    float4* op = reinterpret_cast<float4*>(out) + b * 4;
#pragma unroll
    for (int q = 0; q < 4; ++q) op[q] = t4[idx * 4 + q];
}

extern "C" void kernel_launch(void* const* d_in, const int* in_sizes, int n_in,
                              void* d_out, int out_size, void* d_ws, size_t ws_size,
                              hipStream_t stream) {
    const int*   x     = (const int*)  d_in[0];
    const float* emb   = (const float*)d_in[1];
    const float* w1    = (const float*)d_in[2];
    const float* b1    = (const float*)d_in[3];
    // d_in[4] = w2 : unused by the reference computation
    const float* b2    = (const float*)d_in[5];
    const float* w_out = (const float*)d_in[6];
    const float* b_out = (const float*)d_in[7];
    float*       out   = (float*)d_out;
    float*       table = (float*)d_ws;        // 256 floats of scratch

    dclstm_build_table<<<1, 32, 0, stream>>>(emb, w1, b1, b2, w_out, b_out, table);
    dclstm_gather<<<4096 / 256, 256, 0, stream>>>(x, table, out);
}
